// KLDivAndAngleLoss_6107443495113
// MI455X (gfx1250) — matrix operand order repaired and checked by reading, not compiled, not run on hardware
//
#include <hip/hip_runtime.h>
#include <hip/hip_bf16.h>

// KLDivAndAngleLoss for MI455X (gfx1250).
// Key transform: angle tensor e[b,i,j,k] = <d_ij, d_ik> with d_ij = normalize(x_i - x_j)
// collapses to (1 - S[i,j] - S[i,k] + S[j,k]) / (r_ij * r_ik), S = X X^T (unit rows),
// r_ij = sqrt(2 - 2 S[i,j]).  S computed with v_wmma_f32_16x16x32_f16 using an
// f16 hi/lo split (hi*hi + hi*lo + lo*hi) for ~f32 accuracy.
// B=16, K=128, N=512.  Deterministic two-stage reductions; no atomics.

#define BB 16
#define KK 128
#define NN 512
#define BK (BB * KK)              // 2048 rows
#define SIDE_H (BB * KK * NN)     // 1,048,576 halves per side
#define SIDE_S (BB * KK * KK)     // 262,144 floats per side
#define EPSV 1e-12f

typedef __attribute__((ext_vector_type(16))) _Float16 v16h;
typedef __attribute__((ext_vector_type(8)))  float    v8f;

union PackV16H { uint4 u[2]; v16h v; };

// ---------- block reductions (256 threads, fixed order -> deterministic) ----------
__device__ __forceinline__ float bsum256(float v, float* sh) {
    int tid = threadIdx.x;
    sh[tid] = v; __syncthreads();
#pragma unroll
    for (int s = 128; s >= 1; s >>= 1) {
        if (tid < s) sh[tid] = sh[tid] + sh[tid + s];
        __syncthreads();
    }
    float r = sh[0]; __syncthreads();
    return r;
}

__device__ __forceinline__ float bmax256(float v, float* sh) {
    int tid = threadIdx.x;
    sh[tid] = v; __syncthreads();
#pragma unroll
    for (int s = 128; s >= 1; s >>= 1) {
        if (tid < s) sh[tid] = fmaxf(sh[tid], sh[tid + s]);
        __syncthreads();
    }
    float r = sh[0]; __syncthreads();
    return r;
}

// ---------- 1) L2-normalize rows, split into f16 hi + f16 lo ----------
// grid = 2*BK blocks (side major), 256 threads, 2 elements/thread
__global__ void normalize_split_kernel(const float* __restrict__ xs,
                                       const float* __restrict__ xt,
                                       _Float16* __restrict__ hi,
                                       _Float16* __restrict__ lo) {
    __shared__ float sh[256];
    int side = blockIdx.x >> 11;          // 0 = student, 1 = teacher
    int row  = blockIdx.x & 2047;
    const float* src = (side ? xt : xs) + (size_t)row * NN;
    int tid = threadIdx.x;
    float a = src[tid];
    float c = src[tid + 256];
    float ss  = bsum256(a * a + c * c, sh);
    float inv = 1.0f / fmaxf(sqrtf(ss), EPSV);
    size_t base = (size_t)side * SIDE_H + (size_t)row * NN;
    float y0 = a * inv; _Float16 h0 = (_Float16)y0;
    hi[base + tid]       = h0;
    lo[base + tid]       = (_Float16)(y0 - (float)h0);
    float y1 = c * inv; _Float16 h1 = (_Float16)y1;
    hi[base + 256 + tid] = h1;
    lo[base + 256 + tid] = (_Float16)(y1 - (float)h1);
}

// ---------- 2) Per-batch Gram S = Xn * Xn^T via WMMA (hi/lo split) ----------
// grid = 2 sides * 16 b * 8 mt * 8 nt = 2048 blocks, 32 threads (1 wave, EXEC all-1s)
__global__ void gram_wmma_kernel(const _Float16* __restrict__ hi,
                                 const _Float16* __restrict__ lo,
                                 float* __restrict__ S) {
    int id   = blockIdx.x;
    int nt   = id & 7;  id >>= 3;
    int mt   = id & 7;  id >>= 3;
    int b    = id & 15;
    int side = id >> 4;

    const _Float16* hiP = hi + (size_t)side * SIDE_H + (size_t)b * (KK * NN);
    const _Float16* loP = lo + (size_t)side * SIDE_H + (size_t)b * (KK * NN);

    int lane = threadIdx.x;
    int m    = lane & 15;
    int hsel = lane >> 4;

    // A operand: row (mt*16 + m), two contiguous 8-half chunks per 32-K step
    const _Float16* rowAhi = hiP + (size_t)(mt * 16 + m) * NN;
    const _Float16* rowAlo = loP + (size_t)(mt * 16 + m) * NN;
    // B operand: columns of rows nt*16 .. nt*16+15, lane m indexes N, elements index K
    const _Float16* bh = hiP + (size_t)(nt * 16) * NN;
    const _Float16* bl = loP + (size_t)(nt * 16) * NN;

    v8f acc = {};
    for (int k0 = 0; k0 < NN; k0 += 32) {
        PackV16H pa_hi, pa_lo;
        pa_hi.u[0] = *(const uint4*)(rowAhi + k0 + hsel * 8);
        pa_hi.u[1] = *(const uint4*)(rowAhi + k0 + 16 + hsel * 8);
        pa_lo.u[0] = *(const uint4*)(rowAlo + k0 + hsel * 8);
        pa_lo.u[1] = *(const uint4*)(rowAlo + k0 + 16 + hsel * 8);

        int kcol = k0 + hsel * 8;
        v16h vb_hi, vb_lo;
#pragma unroll
        for (int e = 0; e < 16; ++e) {
            vb_hi[e] = bh[m * NN + kcol + (e & 7) + (e >> 3) * 16];
            vb_lo[e] = bl[m * NN + kcol + (e & 7) + (e >> 3) * 16];
        }

        acc = __builtin_amdgcn_wmma_f32_16x16x32_f16(false, pa_hi.v, false, vb_hi,
                                                     (short)0, acc, false, false);
        acc = __builtin_amdgcn_wmma_f32_16x16x32_f16(false, pa_hi.v, false, vb_lo,
                                                     (short)0, acc, false, false);
        acc = __builtin_amdgcn_wmma_f32_16x16x32_f16(false, pa_lo.v, false, vb_hi,
                                                     (short)0, acc, false, false);
    }

    float* Sp = S + (size_t)side * SIDE_S + (size_t)b * (KK * KK);
    int col = nt * 16 + m;
#pragma unroll
    for (int r = 0; r < 8; ++r) {
        int row = mt * 16 + r + 8 * hsel;
        Sp[row * KK + col] = acc[r];
    }
}

// ---------- 3) KL over class dim per (b,k) row ----------
// grid = BK blocks, 256 threads, 2 elements/thread
__global__ void kl_kernel(const float* __restrict__ xs,
                          const float* __restrict__ xt,
                          float* __restrict__ pkl) {
    __shared__ float sh[256];
    int row = blockIdx.x;
    int tid = threadIdx.x;
    const float* s = xs + (size_t)row * NN;
    const float* t = xt + (size_t)row * NN;
    float s0 = s[tid], s1 = s[tid + 256];
    float t0 = t[tid], t1 = t[tid + 256];

    float invs = 1.0f / fmaxf(sqrtf(bsum256(s0 * s0 + s1 * s1, sh)), EPSV);
    float invt = 1.0f / fmaxf(sqrtf(bsum256(t0 * t0 + t1 * t1, sh)), EPSV);
    float ys0 = s0 * invs, ys1 = s1 * invs;
    float yt0 = t0 * invt, yt1 = t1 * invt;

    float ms   = bmax256(fmaxf(ys0, ys1), sh);
    float lses = logf(bsum256(expf(ys0 - ms) + expf(ys1 - ms), sh));
    float mt_  = bmax256(fmaxf(yt0, yt1), sh);
    float lset = logf(bsum256(expf(yt0 - mt_) + expf(yt1 - mt_), sh));

    float lp0 = yt0 - mt_ - lset, lp1 = yt1 - mt_ - lset;   // log_pt
    float lq0 = ys0 - ms - lses,  lq1 = ys1 - ms - lses;    // log_ps
    float c = expf(lp0) * (lp0 - lq0) + expf(lp1) * (lp1 - lq1);
    float tot = bsum256(c, sh);
    if (tid == 0) pkl[row] = tot;
}

// ---------- 4) Angle smooth-L1 from Grams ----------
// grid = B*K blocks (one per (b, anchor i)), 256 threads
__global__ void angle_kernel(const float* __restrict__ S,
                             float* __restrict__ pang) {
    __shared__ float sh[256];
    __shared__ float rs[KK], rt[KK], is_[KK], it_[KK];
    int blk = blockIdx.x;
    int b = blk >> 7, i = blk & 127;
    const float* Ss = S + (size_t)b * (KK * KK);
    const float* St = S + SIDE_S + (size_t)b * (KK * KK);
    int tid = threadIdx.x;
    if (tid < KK) {
        float vs = Ss[i * KK + tid];
        float vt = St[i * KK + tid];
        rs[tid] = vs; rt[tid] = vt;
        is_[tid] = 1.0f / fmaxf(sqrtf(fmaxf(2.0f - 2.0f * vs, 0.0f)), EPSV);
        it_[tid] = 1.0f / fmaxf(sqrtf(fmaxf(2.0f - 2.0f * vt, 0.0f)), EPSV);
    }
    __syncthreads();
    float loss = 0.0f;
    for (int idx = tid; idx < KK * KK; idx += 256) {
        int j = idx >> 7, k = idx & 127;
        if (j != i && k != i) {  // d_ii == 0 exactly in reference -> e == 0
            float es = (1.0f - rs[j] - rs[k] + Ss[j * KK + k]) * is_[j] * is_[k];
            float et = (1.0f - rt[j] - rt[k] + St[j * KK + k]) * it_[j] * it_[k];
            float d = es - et, ad = fabsf(d);
            loss += (ad < 1.0f) ? 0.5f * d * d : (ad - 0.5f);
        }
    }
    float tot = bsum256(loss, sh);
    if (tid == 0) pang[blk] = tot;
}

// ---------- 5) deterministic final reduction ----------
__global__ void finalize_kernel(const float* __restrict__ pkl,
                                const float* __restrict__ pang,
                                float* __restrict__ out) {
    __shared__ float sh[256];
    int tid = threadIdx.x;
    float a = 0.0f, g = 0.0f;
    for (int i = tid; i < BK; i += 256) { a += pkl[i]; g += pang[i]; }
    float kl  = bsum256(a, sh);
    float ang = bsum256(g, sh);
    if (tid == 0)
        out[0] = kl / (float)BK + ang / 33554432.0f;  // B*K^3 = 16*128^3
}

extern "C" void kernel_launch(void* const* d_in, const int* in_sizes, int n_in,
                              void* d_out, int out_size, void* d_ws, size_t ws_size,
                              hipStream_t stream) {
    (void)in_sizes; (void)n_in; (void)out_size; (void)ws_size;
    const float* xs = (const float*)d_in[0];   // logits_student [16,128,512] f32
    const float* xt = (const float*)d_in[1];   // logits_teacher [16,128,512] f32
    float* out = (float*)d_out;                // scalar

    // workspace layout (~10.5 MB total)
    _Float16* hi = (_Float16*)d_ws;            // 2 * 1,048,576 halves (4 MB)
    _Float16* lo = hi + 2 * SIDE_H;            // 4 MB
    float*    S  = (float*)(lo + 2 * SIDE_H);  // 2 * 262,144 f32 (2 MB)
    float*    pkl  = S + 2 * SIDE_S;           // 2048 f32
    float*    pang = pkl + BK;                 // 2048 f32

    normalize_split_kernel<<<2 * BK, 256, 0, stream>>>(xs, xt, hi, lo);
    gram_wmma_kernel<<<2 * BB * 8 * 8, 32, 0, stream>>>(hi, lo, S);
    kl_kernel<<<BK, 256, 0, stream>>>(xs, xt, pkl);
    angle_kernel<<<BK, 256, 0, stream>>>(S, pang);
    finalize_kernel<<<1, 256, 0, stream>>>(pkl, pang, out);
}
